// CrossAttentionModule_71751723647778
// MI455X (gfx1250) — matrix operand order repaired and checked by reading, not compiled, not run on hardware
//
#include <hip/hip_runtime.h>

typedef __attribute__((ext_vector_type(16))) _Float16 v16h;
typedef __attribute__((ext_vector_type(8)))  float    v8f;
typedef __attribute__((ext_vector_type(4)))  _Float16 v4h;
typedef __attribute__((ext_vector_type(4)))  unsigned int v4u;

__device__ __forceinline__ float wred_sum(float x) {
#pragma unroll
  for (int o = 16; o > 0; o >>= 1) x += __shfl_xor(x, o, 32);
  return x;
}
__device__ __forceinline__ float wred_max(float x) {
#pragma unroll
  for (int o = 16; o > 0; o >>= 1) x = fmaxf(x, __shfl_xor(x, o, 32));
  return x;
}

// --- CDNA5 async global->LDS copies (ASYNCcnt-tracked, no VGPR staging) -----
__device__ __forceinline__ void async_b128(unsigned lds, const void* g) {
  asm volatile("global_load_async_to_lds_b128 %0, %1, off"
               :: "v"(lds), "v"(g) : "memory");
}
__device__ __forceinline__ void async_b128_o16(unsigned lds, const void* g) {
  asm volatile("global_load_async_to_lds_b128 %0, %1, off offset:16"
               :: "v"(lds), "v"(g) : "memory");
}
__device__ __forceinline__ void wait_async0() {
  asm volatile("s_wait_asynccnt 0" ::: "memory");
}

// ------------- weight convert + transpose: Wt[n][k] = (f16) W[k][n] ---------
__global__ __launch_bounds__(256) void k_cvtT(const float* __restrict__ W,
                                              _Float16* __restrict__ Wt,
                                              int K, int N) {
  int t = blockIdx.x * 256 + threadIdx.x;
  if (t < K * N) {
    int n = t / K, kk = t - n * K;
    Wt[t] = (_Float16)W[(size_t)kk * N + n];
  }
}

// ------------- LayerNorm over last dim 128, one wave per row ----------------
__global__ __launch_bounds__(256) void k_ln128(const float* __restrict__ x,
    const float* __restrict__ g, const float* __restrict__ b,
    float* __restrict__ y32, _Float16* __restrict__ y16, int rows) {
  int w = blockIdx.x * 8 + (threadIdx.x >> 5);
  int lane = threadIdx.x & 31;
  if (w >= rows) return;
  float4 xv = *(const float4*)(x + (size_t)w * 128 + lane * 4);
  float xa[4] = {xv.x, xv.y, xv.z, xv.w};
  float mean = wred_sum(xa[0] + xa[1] + xa[2] + xa[3]) * (1.f / 128.f);
  float vs = 0.f;
#pragma unroll
  for (int q = 0; q < 4; ++q) { float d = xa[q] - mean; vs += d * d; }
  float inv = rsqrtf(wred_sum(vs) * (1.f / 128.f) + 1e-5f);
  float r[4];
#pragma unroll
  for (int q = 0; q < 4; ++q) {
    int d = lane * 4 + q;
    r[q] = (xa[q] - mean) * inv * g[d] + b[d];
  }
  if (y32) { float4 o = {r[0], r[1], r[2], r[3]};
             *(float4*)(y32 + (size_t)w * 128 + lane * 4) = o; }
  if (y16) { v4h h = {(_Float16)r[0], (_Float16)r[1], (_Float16)r[2], (_Float16)r[3]};
             *(v4h*)(y16 + (size_t)w * 128 + lane * 4) = h; }
}

// ------------- gated pair bias (+ mask bias), one wave per pair row ---------
// out[((b*4+h)*NI + i)*NJ + j];  p-orientation: i over NP, j over NC (isC=0)
// c-orientation: i over NC, j over NP, pair row transposed (isC=1)
__global__ __launch_bounds__(256) void k_pair_bias(
    const float* __restrict__ pair, const int* __restrict__ pmask,
    const float* __restrict__ lng, const float* __restrict__ lnb,
    const float* __restrict__ wl, const float* __restrict__ bl,
    const float* __restrict__ wg, const float* __restrict__ bg,
    float* __restrict__ out, int NI, int NJ, int isC) {
  __shared__ float sWl[512], sWg[512], sG[128], sB[128], sBl[4], sBg[4];
  int tid = threadIdx.x;
  for (int t = tid; t < 512; t += 256) { sWl[t] = wl[t]; sWg[t] = wg[t]; }
  if (tid < 128) { sG[tid] = lng[tid]; sB[tid] = lnb[tid]; }
  if (tid < 4) { sBl[tid] = bl[tid]; sBg[tid] = bg[tid]; }
  __syncthreads();
  int w = blockIdx.x * 8 + (tid >> 5);
  int lane = tid & 31;
  int j = w % NJ; int t2 = w / NJ; int i = t2 % NI; int b = t2 / NI;
  size_t prow = isC ? ((size_t)(b * 512 + j) * 128 + i)
                    : ((size_t)(b * 512 + i) * 128 + j);
  float4 xv = *(const float4*)(pair + prow * 128 + lane * 4);
  float xa[4] = {xv.x, xv.y, xv.z, xv.w};
  float mean = wred_sum(xa[0] + xa[1] + xa[2] + xa[3]) * (1.f / 128.f);
  float vs = 0.f;
#pragma unroll
  for (int q = 0; q < 4; ++q) { float d = xa[q] - mean; vs += d * d; }
  float inv = rsqrtf(wred_sum(vs) * (1.f / 128.f) + 1e-5f);
  float pl[4] = {0, 0, 0, 0}, pg[4] = {0, 0, 0, 0};
#pragma unroll
  for (int q = 0; q < 4; ++q) {
    int d = lane * 4 + q;
    float pr = (xa[q] - mean) * inv * sG[d] + sB[d];
#pragma unroll
    for (int h = 0; h < 4; ++h) {
      pl[h] += pr * sWl[d * 4 + h];
      pg[h] += pr * sWg[d * 4 + h];
    }
  }
#pragma unroll
  for (int h = 0; h < 4; ++h) { pl[h] = wred_sum(pl[h]); pg[h] = wred_sum(pg[h]); }
  float mb = pmask[prow] ? 0.f : -1e9f;
  if (lane < 4) {
    float l = pl[lane] + sBl[lane];
    float gg = pg[lane] + sBg[lane];
    float pb = l * (1.f / (1.f + __expf(-gg))) + mb;
    out[(((size_t)b * 4 + lane) * NI + i) * NJ + j] = pb;
  }
}

// ------------- tiled WMMA GEMM: C[M,N] = act(A[M,K] * Bt[N,K]^T + bias) -----
// Double-buffered LDS, tiles staged with global_load_async_to_lds_b128 and
// retired with s_wait_asynccnt; compute = v_wmma_f32_16x16x32_f16.
template <bool HASB, bool LEAKY, bool OUT16>
__global__ __launch_bounds__(256) void k_gemm(
    const _Float16* __restrict__ A, const _Float16* __restrict__ Bt,
    const float* __restrict__ bias, float* __restrict__ C,
    _Float16* __restrict__ C16, int M, int N, int K) {
  constexpr int BM = 64, BN = 64, BK = 64, LDT = BK + 8;
  __shared__ alignas(16) _Float16 As[2][BM][LDT];
  __shared__ alignas(16) _Float16 Bs[2][BN][LDT];
  const int tid = threadIdx.x;
  const int lane = tid & 31, wave = tid >> 5;
  const int lm = lane & 15, lh = lane >> 4;
  const int nTN = N / BN;
  const int m0 = (blockIdx.x / nTN) * BM;
  const int n0 = (blockIdx.x % nTN) * BN;
  const int wm = (wave & 3) * 16;   // 4 row-groups of 16
  const int wn = (wave >> 2) * 32;  // 2 col-groups of 32
  v8f acc0 = {}; v8f acc1 = {};
  const int lr = tid >> 2;          // 0..63 tile row
  const int lc = (tid & 3) * 16;    // 0,16,32,48 tile col (halves)

  auto stage = [&](int buf, int k0) {
    const void* gA = (const void*)(A + (size_t)(m0 + lr) * K + k0 + lc);
    const void* gB = (const void*)(Bt + (size_t)(n0 + lr) * K + k0 + lc);
    unsigned la = (unsigned)(uintptr_t)&As[buf][lr][lc];
    unsigned lb = (unsigned)(uintptr_t)&Bs[buf][lr][lc];
    async_b128(la, gA);
    async_b128_o16(la, gA);
    async_b128(lb, gB);
    async_b128_o16(lb, gB);
  };

  stage(0, 0);
  wait_async0();
  __syncthreads();
  int buf = 0;
  for (int k0 = 0; k0 < K; k0 += BK) {
    const int nk0 = k0 + BK;
    if (nk0 < K) stage(buf ^ 1, nk0);
#pragma unroll
    for (int kk = 0; kk < BK; kk += 32) {
      v16h af, bf0, bf1;
      const _Float16* ap = &As[buf][wm + lm][kk + lh * 16];
      ((v4u*)&af)[0] = *(const v4u*)(ap);
      ((v4u*)&af)[1] = *(const v4u*)(ap + 8);
      const _Float16* bp0 = &Bs[buf][wn + lm][kk + lh * 16];
      ((v4u*)&bf0)[0] = *(const v4u*)(bp0);
      ((v4u*)&bf0)[1] = *(const v4u*)(bp0 + 8);
      const _Float16* bp1 = &Bs[buf][wn + 16 + lm][kk + lh * 16];
      ((v4u*)&bf1)[0] = *(const v4u*)(bp1);
      ((v4u*)&bf1)[1] = *(const v4u*)(bp1 + 8);
      acc0 = __builtin_amdgcn_wmma_f32_16x16x32_f16(false, af, false, bf0,
                                                    (short)0, acc0, false, false);
      acc1 = __builtin_amdgcn_wmma_f32_16x16x32_f16(false, af, false, bf1,
                                                    (short)0, acc1, false, false);
    }
    if (nk0 < K) wait_async0();
    __syncthreads();
    buf ^= 1;
  }
  const int mr = m0 + wm + lh * 8;
  const int nc0 = n0 + wn + lm;
  const int nc1 = nc0 + 16;
  float bv0 = 0.f, bv1 = 0.f;
  if (HASB) { bv0 = bias[nc0]; bv1 = bias[nc1]; }
#pragma unroll
  for (int r = 0; r < 8; ++r) {
    float x0 = acc0[r] + bv0;
    float x1 = acc1[r] + bv1;
    if (LEAKY) {
      x0 = x0 > 0.f ? x0 : 0.01f * x0;
      x1 = x1 > 0.f ? x1 : 0.01f * x1;
    }
    size_t ro = (size_t)(mr + r) * N;
    if (OUT16) { C16[ro + nc0] = (_Float16)x0; C16[ro + nc1] = (_Float16)x1; }
    else       { C[ro + nc0] = x0;             C[ro + nc1] = x1; }
  }
}

// ------------- softmax attention: one wave per (b, head, query) -------------
__global__ __launch_bounds__(256) void k_attn(
    const float* __restrict__ q, const float* __restrict__ k,
    const float* __restrict__ v, const float* __restrict__ bias,
    float* __restrict__ o, int NQ, int NK) {
  __shared__ float sQ[8][32];
  __shared__ float sP[8][512];
  int wave = threadIdx.x >> 5, lane = threadIdx.x & 31;
  int w = blockIdx.x * 8 + wave;
  int qi = w % NQ; int t = w / NQ; int h = t & 3; int b = t >> 2;
  sQ[wave][lane] = q[(size_t)(b * NQ + qi) * 128 + h * 32 + lane] * 0.17677669529f;
  __syncthreads();
  int nk = NK >> 5;
  float sc[16];
  const float* kb = k + (size_t)b * NK * 128 + h * 32;
  const float* bb = bias + (((size_t)b * 4 + h) * NQ + qi) * NK;
  for (int t2 = 0; t2 < nk; ++t2) {
    int key = lane + (t2 << 5);
    const float* kr = kb + (size_t)key * 128;
    float s = 0.f;
#pragma unroll
    for (int c = 0; c < 32; ++c) s += sQ[wave][c] * kr[c];
    sc[t2] = s + bb[key];
  }
  float mx = -1e30f;
  for (int t2 = 0; t2 < nk; ++t2) mx = fmaxf(mx, sc[t2]);
  mx = wred_max(mx);
  float sum = 0.f;
  for (int t2 = 0; t2 < nk; ++t2) { sc[t2] = __expf(sc[t2] - mx); sum += sc[t2]; }
  sum = wred_sum(sum);
  float invs = 1.f / sum;
  for (int t2 = 0; t2 < nk; ++t2) sP[wave][lane + (t2 << 5)] = sc[t2] * invs;
  __syncthreads();
  float oc = 0.f;
  const float* vb = v + (size_t)b * NK * 128 + h * 32 + lane;
  for (int j = 0; j < NK; ++j) oc += sP[wave][j] * vb[(size_t)j * 128];
  o[(size_t)(b * NQ + qi) * 128 + h * 32 + lane] = oc;
}

// ------------- elementwise helpers ------------------------------------------
__global__ __launch_bounds__(256) void k_gate(const float* __restrict__ o,
    const float* __restrict__ g, _Float16* __restrict__ out, int n) {
  int t = blockIdx.x * 256 + threadIdx.x;
  if (t < n) out[t] = (_Float16)(o[t] * (1.f / (1.f + __expf(-g[t]))));
}
__global__ __launch_bounds__(256) void k_resid_mask(const float* __restrict__ base,
    const float* __restrict__ x, const int* __restrict__ mask,
    float* __restrict__ out, int n) {
  int t = blockIdx.x * 256 + threadIdx.x;
  if (t < n) out[t] = base[t] + (mask[t >> 7] ? x[t] : 0.f);
}
__global__ __launch_bounds__(256) void k_add(const float* __restrict__ a,
    const float* __restrict__ b, float* __restrict__ out, int n) {
  int t = blockIdx.x * 256 + threadIdx.x;
  if (t < n) out[t] = a[t] + b[t];
}
__global__ __launch_bounds__(256) void k_mask_pair(float* __restrict__ out,
    const int* __restrict__ mask, size_t n) {
  size_t t = (size_t)blockIdx.x * 256 + threadIdx.x;
  if (t < n) out[t] = mask[t >> 7] ? out[t] : 0.f;
}

// ------------- pair = pair_embed + pp[i] + cc[j]; LN; write f16 -------------
__global__ __launch_bounds__(256) void k_pair_sum_ln(
    const float* __restrict__ pair, const float* __restrict__ pp,
    const float* __restrict__ cc, const float* __restrict__ lng,
    const float* __restrict__ lnb, _Float16* __restrict__ out) {
  __shared__ float sG[128], sB[128];
  int tid = threadIdx.x;
  if (tid < 128) { sG[tid] = lng[tid]; sB[tid] = lnb[tid]; }
  __syncthreads();
  int w = blockIdx.x * 8 + (tid >> 5);
  int lane = tid & 31;
  int j = w % 128; int t2 = w / 128; int i = t2 % 512; int b = t2 / 512;
  float4 xv = *(const float4*)(pair + (size_t)w * 128 + lane * 4);
  float4 pv = *(const float4*)(pp + ((size_t)(b * 512 + i) * 128) + lane * 4);
  float4 cv = *(const float4*)(cc + ((size_t)(b * 128 + j) * 128) + lane * 4);
  float xa[4] = {xv.x + pv.x + cv.x, xv.y + pv.y + cv.y,
                 xv.z + pv.z + cv.z, xv.w + pv.w + cv.w};
  float mean = wred_sum(xa[0] + xa[1] + xa[2] + xa[3]) * (1.f / 128.f);
  float vs = 0.f;
#pragma unroll
  for (int q = 0; q < 4; ++q) { float d = xa[q] - mean; vs += d * d; }
  float inv = rsqrtf(wred_sum(vs) * (1.f / 128.f) + 1e-5f);
  v4h hv;
#pragma unroll
  for (int q = 0; q < 4; ++q) {
    int d = lane * 4 + q;
    hv[q] = (_Float16)((xa[q] - mean) * inv * sG[d] + sB[d]);
  }
  *(v4h*)(out + (size_t)w * 128 + lane * 4) = hv;
}

// ============================ host orchestration ============================
extern "C" void kernel_launch(void* const* d_in, const int* in_sizes, int n_in,
                              void* d_out, int out_size, void* d_ws, size_t ws_size,
                              hipStream_t stream) {
  (void)in_sizes; (void)n_in; (void)out_size; (void)ws_size;
  const int B = 4, NP = 512, NC = 128;
  const int MP = B * NP, MC = B * NC;
  const size_t MPAIR = (size_t)B * NP * NC;

  const float* p_embed  = (const float*)d_in[0];
  const float* c_embed  = (const float*)d_in[1];
  const float* pair     = (const float*)d_in[2];
  const int*   p_mask   = (const int*)d_in[3];
  const int*   c_mask   = (const int*)d_in[4];
  const int*   pair_msk = (const int*)d_in[5];
  auto F = [&](int i) { return (const float*)d_in[i]; };
  // p_blk: 6..22, c_blk: 23..39, p_tr: 40..45, c_tr: 46..51,
  // pair_tr: 52..57, inter: 58..65

  char* cur = (char*)d_ws;
  auto alloc = [&](size_t bytes) -> void* {
    void* p = (void*)cur; cur += (bytes + 255) & ~(size_t)255; return p;
  };
  auto cvtT = [&](const float* W, int K, int N) -> _Float16* {
    _Float16* Wt = (_Float16*)alloc((size_t)K * N * sizeof(_Float16));
    k_cvtT<<<dim3((K * N + 255) / 256), dim3(256), 0, stream>>>(W, Wt, K, N);
    return Wt;
  };
  auto gemm = [&](const _Float16* A, const _Float16* Bt, const float* bias,
                  float* C, _Float16* C16, int M, int N, int K, bool leaky) {
    dim3 grid((M / 64) * (N / 64)), blk(256);
    if (C16)
      k_gemm<true, true, true><<<grid, blk, 0, stream>>>(A, Bt, bias, nullptr, C16, M, N, K);
    else if (bias && leaky)
      k_gemm<true, true, false><<<grid, blk, 0, stream>>>(A, Bt, bias, C, nullptr, M, N, K);
    else if (bias)
      k_gemm<true, false, false><<<grid, blk, 0, stream>>>(A, Bt, bias, C, nullptr, M, N, K);
    else
      k_gemm<false, false, false><<<grid, blk, 0, stream>>>(A, Bt, nullptr, C, nullptr, M, N, K);
  };

  // ---- f16 transposed weights ----
  _Float16 *wqT_p = cvtT(F(16), 128, 128), *wkT_p = cvtT(F(17), 128, 128),
           *wvT_p = cvtT(F(18), 128, 128), *wgoT_p = cvtT(F(19), 128, 128),
           *woT_p = cvtT(F(21), 128, 128);
  _Float16 *wqT_c = cvtT(F(33), 128, 128), *wkT_c = cvtT(F(34), 128, 128),
           *wvT_c = cvtT(F(35), 128, 128), *wgoT_c = cvtT(F(36), 128, 128),
           *woT_c = cvtT(F(38), 128, 128);
  _Float16 *w1T_p = cvtT(F(42), 128, 512), *w2T_p = cvtT(F(44), 512, 128);
  _Float16 *w1T_c = cvtT(F(48), 128, 512), *w2T_c = cvtT(F(50), 512, 128);
  _Float16 *w1T_r = cvtT(F(54), 128, 512), *w2T_r = cvtT(F(56), 512, 128);
  _Float16 *wpT   = cvtT(F(62), 128, 128), *wcT = cvtT(F(64), 128, 128);

  // ---- workspace buffers ----
  float*    ni_p  = (float*)alloc((size_t)MP * 128 * 4);
  _Float16* nip16 = (_Float16*)alloc((size_t)MP * 128 * 2);
  _Float16* njc16 = (_Float16*)alloc((size_t)MC * 128 * 2);
  float*    biasP = (float*)alloc((size_t)B * 4 * NP * NC * 4);
  float*    q_p   = (float*)alloc((size_t)MP * 128 * 4);
  float*    k_cb  = (float*)alloc((size_t)MC * 128 * 4);
  float*    v_cb  = (float*)alloc((size_t)MC * 128 * 4);
  float*    g_p   = (float*)alloc((size_t)MP * 128 * 4);
  float*    o_p   = (float*)alloc((size_t)MP * 128 * 4);
  _Float16* og16p = (_Float16*)alloc((size_t)MP * 128 * 2);
  float*    oproj = (float*)alloc((size_t)MP * 128 * 4);
  float*    p1    = (float*)alloc((size_t)MP * 128 * 4);
  float*    ni_c  = (float*)alloc((size_t)MC * 128 * 4);
  _Float16* nic16 = (_Float16*)alloc((size_t)MC * 128 * 2);
  _Float16* njp16 = (_Float16*)alloc((size_t)MP * 128 * 2);
  float*    biasC = (float*)alloc((size_t)B * 4 * NC * NP * 4);
  float*    q_c   = (float*)alloc((size_t)MC * 128 * 4);
  float*    k_pb  = (float*)alloc((size_t)MP * 128 * 4);
  float*    v_pb  = (float*)alloc((size_t)MP * 128 * 4);
  float*    g_c   = (float*)alloc((size_t)MC * 128 * 4);
  float*    o_c   = (float*)alloc((size_t)MC * 128 * 4);
  _Float16* og16c = (_Float16*)alloc((size_t)MC * 128 * 2);
  float*    projc = (float*)alloc((size_t)MC * 128 * 4);
  float*    c1    = (float*)alloc((size_t)MC * 128 * 4);
  _Float16* xt16  = (_Float16*)alloc((size_t)MP * 128 * 2);
  _Float16* h16   = (_Float16*)alloc((size_t)MP * 512 * 2);
  float*    yt    = (float*)alloc((size_t)MP * 128 * 4);
  _Float16* lnp16 = (_Float16*)alloc((size_t)MP * 128 * 2);
  _Float16* lnc16 = (_Float16*)alloc((size_t)MC * 128 * 2);
  float*    ppb   = (float*)alloc((size_t)MP * 128 * 4);
  float*    ccb   = (float*)alloc((size_t)MC * 128 * 4);
  _Float16* pairh = (_Float16*)alloc(MPAIR * 128 * 2);
  const int CHR   = 16384;
  _Float16* h1    = (_Float16*)alloc((size_t)CHR * 512 * 2);

  float* outP = (float*)d_out;
  float* outC = outP + (size_t)MP * 128;
  float* outPair = outC + (size_t)MC * 128;

  // ================= protein row-attention over compound keys ================
  k_ln128<<<dim3(MP / 8), 256, 0, stream>>>(p_embed, F(6), F(7), ni_p, nip16, MP);
  k_ln128<<<dim3(MC / 8), 256, 0, stream>>>(c_embed, F(8), F(9), nullptr, njc16, MC);
  k_pair_bias<<<dim3((int)(MPAIR / 8)), 256, 0, stream>>>(
      pair, pair_msk, F(10), F(11), F(12), F(13), F(14), F(15), biasP, NP, NC, 0);
  gemm(nip16, wqT_p, nullptr, q_p, nullptr, MP, 128, 128, false);
  gemm(njc16, wkT_p, nullptr, k_cb, nullptr, MC, 128, 128, false);
  gemm(njc16, wvT_p, nullptr, v_cb, nullptr, MC, 128, 128, false);
  gemm(nip16, wgoT_p, F(20), g_p, nullptr, MP, 128, 128, false);
  k_attn<<<dim3(B * 4 * NP / 8), 256, 0, stream>>>(q_p, k_cb, v_cb, biasP, o_p, NP, NC);
  k_gate<<<dim3(MP * 128 / 256), 256, 0, stream>>>(o_p, g_p, og16p, MP * 128);
  gemm(og16p, woT_p, F(22), oproj, nullptr, MP, 128, 128, false);
  k_resid_mask<<<dim3(MP * 128 / 256), 256, 0, stream>>>(ni_p, oproj, p_mask, p1, MP * 128);

  // ================= compound row-attention over protein keys ================
  k_ln128<<<dim3(MC / 8), 256, 0, stream>>>(c_embed, F(23), F(24), ni_c, nic16, MC);
  k_ln128<<<dim3(MP / 8), 256, 0, stream>>>(p1, F(25), F(26), nullptr, njp16, MP);
  k_pair_bias<<<dim3((int)(MPAIR / 8)), 256, 0, stream>>>(
      pair, pair_msk, F(27), F(28), F(29), F(30), F(31), F(32), biasC, NC, NP, 1);
  gemm(nic16, wqT_c, nullptr, q_c, nullptr, MC, 128, 128, false);
  gemm(njp16, wkT_c, nullptr, k_pb, nullptr, MP, 128, 128, false);
  gemm(njp16, wvT_c, nullptr, v_pb, nullptr, MP, 128, 128, false);
  gemm(nic16, wgoT_c, F(37), g_c, nullptr, MC, 128, 128, false);
  k_attn<<<dim3(B * 4 * NC / 8), 256, 0, stream>>>(q_c, k_pb, v_pb, biasC, o_c, NC, NP);
  k_gate<<<dim3(MC * 128 / 256), 256, 0, stream>>>(o_c, g_c, og16c, MC * 128);
  gemm(og16c, woT_c, F(39), projc, nullptr, MC, 128, 128, false);
  k_resid_mask<<<dim3(MC * 128 / 256), 256, 0, stream>>>(ni_c, projc, c_mask, c1, MC * 128);

  // ================= node transitions (residual MLPs) ========================
  k_ln128<<<dim3(MP / 8), 256, 0, stream>>>(p1, F(40), F(41), nullptr, xt16, MP);
  gemm(xt16, w1T_p, F(43), nullptr, h16, MP, 512, 128, true);
  gemm(h16, w2T_p, F(45), yt, nullptr, MP, 128, 512, true);
  k_add<<<dim3(MP * 128 / 256), 256, 0, stream>>>(p1, yt, outP, MP * 128);

  k_ln128<<<dim3(MC / 8), 256, 0, stream>>>(c1, F(46), F(47), nullptr, xt16, MC);
  gemm(xt16, w1T_c, F(49), nullptr, h16, MC, 512, 128, true);
  gemm(h16, w2T_c, F(51), yt, nullptr, MC, 128, 512, true);
  k_add<<<dim3(MC * 128 / 256), 256, 0, stream>>>(c1, yt, outC, MC * 128);

  // ================= interaction + pair transition ===========================
  k_ln128<<<dim3(MP / 8), 256, 0, stream>>>(outP, F(58), F(59), nullptr, lnp16, MP);
  gemm(lnp16, wpT, F(63), ppb, nullptr, MP, 128, 128, false);
  k_ln128<<<dim3(MC / 8), 256, 0, stream>>>(outC, F(60), F(61), nullptr, lnc16, MC);
  gemm(lnc16, wcT, F(65), ccb, nullptr, MC, 128, 128, false);
  k_pair_sum_ln<<<dim3((int)(MPAIR / 8)), 256, 0, stream>>>(
      pair, ppb, ccb, F(52), F(53), pairh);
  for (size_t c0 = 0; c0 < MPAIR; c0 += CHR) {
    gemm(pairh + c0 * 128, w1T_r, F(55), nullptr, h1, CHR, 512, 128, true);
    gemm(h1, w2T_r, F(57), outPair + c0 * 128, nullptr, CHR, 128, 512, true);
  }
  k_mask_pair<<<dim3((int)((MPAIR * 128 + 255) / 256)), 256, 0, stream>>>(
      outPair, pair_msk, MPAIR * 128);
}
